// EquivariantEdgeConv_57269093925337
// MI455X (gfx1250) — matrix operand order repaired and checked by reading, not compiled, not run on hardware
//
#include <hip/hip_runtime.h>
#include <hip/hip_bf16.h>
#include <math.h>

typedef __attribute__((ext_vector_type(16))) _Float16 v16h;
typedef __attribute__((ext_vector_type(8)))  float    v8f;

#define NNODES 8192
#define NEDGES 65536
#define Hc 64
#define WNUMc 4096

__device__ __forceinline__ float sig_(float z) { return 1.0f / (1.0f + __expf(-z)); }

// ---------------------------------------------------------------------------
// Kernel 0: repack w2 (f32, row-major [64][4096]) into f16 WMMA-B fragment
// order.  Fragment f = (t*2 + wh)*32 + lane holds 16 halves:
//   element e -> k = wh*32 + ((lane>=16)?16:0) + e ,  n = t*16 + (lane&15)
// ---------------------------------------------------------------------------
__global__ void repack_w2_kernel(const float* __restrict__ w2,
                                 _Float16* __restrict__ w2h) {
    int tid = blockIdx.x * blockDim.x + threadIdx.x;
    if (tid >= Hc * WNUMc) return;
    int e    = tid & 15;
    int lane = (tid >> 4) & 31;
    int wh   = (tid >> 9) & 1;
    int t    = tid >> 10;
    int k = wh * 32 + ((lane & 16) ? 16 : 0) + e;
    int n = t * 16 + (lane & 15);
    w2h[tid] = (_Float16)w2[k * WNUMc + n];
}

__global__ void zero_kernel(float* __restrict__ p, int n) {
    int i = blockIdx.x * blockDim.x + threadIdx.x;
    if (i < n) p[i] = 0.0f;
}

// ---------------------------------------------------------------------------
// Kernel 1: fused edge kernel. 128 threads = 4 waves, each wave owns 16 edges.
// A fragments (h = 0.125*silu(len*w1)) built in registers; 256 column tiles of
// w2 via v_wmma_f32_16x16x32_f16 (K=64 = 2 WMMAs); tile consumption goes to
// REGISTER accumulators (o depends on t only through t%3 / is constant), LDS
// is only read in the loop; one LDS write pass at the end, then atomic scatter.
// ---------------------------------------------------------------------------
__global__ __launch_bounds__(128) void edge_kernel(
    const float* __restrict__ x, const float* __restrict__ pos,
    const int* __restrict__ ei, const float* __restrict__ w1,
    const _Float16* __restrict__ w2h, float* __restrict__ acc)
{
    __shared__ float smem[4][3648];
    const int lane = threadIdx.x & 31;
    const int wid  = threadIdx.x >> 5;
    float* Wm   = smem[wid];
    float* sY   = Wm;          // [16][4]  Y1.xyz, len
    float* sXS  = Wm + 64;     // [16][48] cA * xs
    float* sXV  = Wm + 832;    // [16][48] cD * xv
    float* sXVY = Wm + 1600;   // [16][16] cB * (xv . Y1)
    float* sMS  = Wm + 1856;   // [16][48] ms staging
    float* sU   = Wm + 2624;   // [16][16] cC*u staging
    float* sMV  = Wm + 2880;   // [16][48] wD-part of mv staging

    const int   e0   = (blockIdx.x * 4 + wid) * 16;
    const float cA   = 0.1020620726f;   // 1/sqrt(96) == cB == cC
    const float cD   = 0.1767766953f;   // 1/sqrt(32)
    const float cBoD = 0.5773502692f;   // cB / cD

    for (int idx = lane; idx < 16 * 96; idx += 32) {
        int e = idx / 96, ch = idx - e * 96;
        int s = ei[e0 + e];
        float v = x[s * 96 + ch];
        if (ch < 48) sXS[e * 48 + ch] = cA * v;
        else         sXV[e * 48 + (ch - 48)] = cD * v;
    }
    if (lane < 16) {
        int ge = e0 + lane;
        int s = ei[ge], d = ei[NEDGES + ge];
        float vx = pos[d * 3 + 0] - pos[s * 3 + 0];
        float vy = pos[d * 3 + 1] - pos[s * 3 + 1];
        float vz = pos[d * 3 + 2] - pos[s * 3 + 2];
        float len = fmaxf(sqrtf(vx * vx + vy * vy + vz * vz), 1e-8f);
        float inv = 1.7320508076f / len;   // sqrt(3)/len
        sY[lane * 4 + 0] = vx * inv;
        sY[lane * 4 + 1] = vy * inv;
        sY[lane * 4 + 2] = vz * inv;
        sY[lane * 4 + 3] = len;
    }
    __syncthreads();
    for (int idx = lane; idx < 256; idx += 32) {
        int e = idx >> 4, i = idx & 15;
        sXVY[idx] = cBoD * (sXV[e * 48 + 3 * i + 0] * sY[e * 4 + 0] +
                            sXV[e * 48 + 3 * i + 1] * sY[e * 4 + 1] +
                            sXV[e * 48 + 3 * i + 2] * sY[e * 4 + 2]);
    }
    __syncthreads();

    // A fragments in contiguous k order (same as w2h repack):
    // lane L row = L&15 ; element e -> K = 16*(L>=16) + e
    const int hi = (lane >> 4) & 1;
    const float len = sY[(lane & 15) * 4 + 3];
    v16h a0, a1;
#pragma unroll
    for (int e2 = 0; e2 < 16; ++e2) {
        int k = (hi ? 16 : 0) + e2;
        float z0 = len * w1[k];
        float z1 = len * w1[k + 32];
        a0[e2] = (_Float16)(0.125f * z0 * sig_(z0));   // 1/sqrt(H) folded in
        a1[e2] = (_Float16)(0.125f * z1 * sig_(z1));
    }

    const v16h* __restrict__ w2v = (const v16h*)w2h;
    const int col   = lane & 15;
    const int ebase = hi * 8;

    auto wmma_tile = [&](int t) -> v8f {
        v16h b0 = w2v[t * 64 + lane];
        v16h b1 = w2v[t * 64 + 32 + lane];
        __builtin_prefetch(&w2v[(t + 1) * 64 + lane], 0, 3);
        v8f c = {};
        c = __builtin_amdgcn_wmma_f32_16x16x32_f16(
            false, a0, false, b0, (short)0, c, false, false);
        c = __builtin_amdgcn_wmma_f32_16x16x32_f16(
            false, a1, false, b1, (short)0, c, false, false);
        return c;
    };

    // Register accumulators: slot s holds output o = (16*s + col) % 48
    float msA[3][8] = {{0.f}};   // ms (wA + wB regions share slot mapping)
    float uA[8]     = {0.f};     // cC*u at o = col
    float mvA[8][3] = {{0.f}};   // wD-part of mv at o = col

    // wA region: tiles 0..143 (multiple of 3 -> slot = t%3 is the unroll index)
    for (int tb = 0; tb < 144; tb += 3) {
#pragma unroll
        for (int s = 0; s < 3; ++s) {
            int t = tb + s;
            v8f c = wmma_tile(t);
            int i = (t * 16 + col) / 48;
#pragma unroll
            for (int v = 0; v < 8; ++v)
                msA[s][v] += sXS[(ebase + v) * 48 + i] * c[v];
        }
    }
    // wB region: tiles 144..191 (144 % 3 == 0, 2304 % 48 == 0 -> same slots)
    for (int tb = 144; tb < 192; tb += 3) {
#pragma unroll
        for (int s = 0; s < 3; ++s) {
            int t = tb + s;
            v8f c = wmma_tile(t);
            int i = (t * 16 + col) / 48 - 48;
#pragma unroll
            for (int v = 0; v < 8; ++v)
                msA[s][v] += sXVY[(ebase + v) * 16 + i] * c[v];
        }
    }
    // wC region: tiles 192..239 ; o == col, i == t - 192
    for (int t = 192; t < 240; ++t) {
        v8f c = wmma_tile(t);
        int i = t - 192;
#pragma unroll
        for (int v = 0; v < 8; ++v)
            uA[v] += sXS[(ebase + v) * 48 + i] * c[v];
    }
    // wD region: tiles 240..255 ; o == col, i == t - 240
    for (int t = 240; t < 256; ++t) {
        v8f c = wmma_tile(t);
        int i = t - 240;
#pragma unroll
        for (int v = 0; v < 8; ++v) {
            float wv = c[v];
            mvA[v][0] += sXV[(ebase + v) * 48 + 3 * i + 0] * wv;
            mvA[v][1] += sXV[(ebase + v) * 48 + 3 * i + 1] * wv;
            mvA[v][2] += sXV[(ebase + v) * 48 + 3 * i + 2] * wv;
        }
    }

    // Stage register accumulators to LDS (all (e,o) targets distinct per lane)
#pragma unroll
    for (int s = 0; s < 3; ++s) {
        int o = (16 * s + col) % 48;
#pragma unroll
        for (int v = 0; v < 8; ++v)
            sMS[(ebase + v) * 48 + o] = msA[s][v];
    }
#pragma unroll
    for (int v = 0; v < 8; ++v) {
        sU[(ebase + v) * 16 + col] = uA[v];
        sMV[(ebase + v) * 48 + 3 * col + 0] = mvA[v][0];
        sMV[(ebase + v) * 48 + 3 * col + 1] = mvA[v][1];
        sMV[(ebase + v) * 48 + 3 * col + 2] = mvA[v][2];
    }
    __syncthreads();

    // finalize message and scatter-add to node accumulator
    for (int idx = lane; idx < 16 * 96; idx += 32) {
        int e = idx / 96, ch = idx - e * 96;
        int d = ei[NEDGES + e0 + e];
        float val;
        if (ch < 48) {
            val = sMS[e * 48 + ch];
        } else {
            int cc = ch - 48;
            int o = cc / 3, m = cc - o * 3;
            val = sMV[e * 48 + cc] + sU[e * 16 + o] * sY[e * 4 + m];
        }
        atomicAdd(&acc[d * 96 + ch], val);
    }
}

// ---------------------------------------------------------------------------
// Kernel 2: node update.  96 threads = 2 nodes x 48 output channels.
// ---------------------------------------------------------------------------
__global__ __launch_bounds__(96) void node_kernel(
    const float* __restrict__ acc, const float* __restrict__ Ws,
    const float* __restrict__ Wns, const float* __restrict__ Wg,
    float* __restrict__ out)
{
    __shared__ float sA[2][96];
    const int tid = threadIdx.x;
    const int nb  = blockIdx.x * 2;
    for (int idx = tid; idx < 192; idx += 96) {
        int ln = idx / 96, ch = idx - ln * 96;
        sA[ln][ch] = acc[(nb + ln) * 96 + ch];
    }
    __syncthreads();

    int ln = tid / 48, ch = tid - ln * 48;
    float as = 0.0f, ag = 0.0f;
#pragma unroll 4
    for (int i = 0; i < 48; ++i) {
        float xv = sA[ln][i];
        as += xv * Ws[i * 48 + ch];
        ag += xv * Wg[i * 48 + ch];
    }
    int o = ch / 3, m = ch - o * 3;
    float an = 0.0f;
#pragma unroll
    for (int i = 0; i < 16; ++i)
        an += sA[ln][48 + 3 * i + m] * Wns[i * 16 + o];

    const float cs = 0.1443375673f;  // 1/sqrt(48)
    float s = as * cs; s = s * sig_(s);
    float g = sig_(ag * cs);
    out[(nb + ln) * 48 + ch] = s + g * an * 0.25f;   // 0.25 = 1/sqrt(16)
}

// ---------------------------------------------------------------------------
extern "C" void kernel_launch(void* const* d_in, const int* in_sizes, int n_in,
                              void* d_out, int out_size, void* d_ws, size_t ws_size,
                              hipStream_t stream) {
    const float* x   = (const float*)d_in[0];
    const float* pos = (const float*)d_in[1];
    const int*   ei  = (const int*)d_in[2];
    const float* w1  = (const float*)d_in[3];
    const float* w2  = (const float*)d_in[4];
    const float* Ws  = (const float*)d_in[5];
    const float* Wns = (const float*)d_in[6];
    const float* Wg  = (const float*)d_in[7];
    float* out = (float*)d_out;

    _Float16* w2h = (_Float16*)d_ws;                                  // 512 KB
    float* acc = (float*)((char*)d_ws + (size_t)Hc * WNUMc * sizeof(_Float16));

    repack_w2_kernel<<<(Hc * WNUMc + 255) / 256, 256, 0, stream>>>(w2, w2h);
    zero_kernel<<<(NNODES * 96 + 255) / 256, 256, 0, stream>>>(acc, NNODES * 96);
    edge_kernel<<<NEDGES / 64, 128, 0, stream>>>(x, pos, ei, w1, w2h, acc);
    node_kernel<<<NNODES / 2, 96, 0, stream>>>(acc, Ws, Wns, Wg, out);
    (void)in_sizes; (void)n_in; (void)out_size; (void)ws_size;
}
